// AttLayer_4105988735233
// MI455X (gfx1250) — compile-verified, matrix operand order checked by reading, not run on hardware
//
#include <hip/hip_runtime.h>
#include <hip/hip_bf16.h>

#define BB  8
#define CC  512
#define HWp 1024
#define BHW (BB * HWp)

typedef _Float16 v16h __attribute__((ext_vector_type(16)));
typedef _Float16 v8h  __attribute__((ext_vector_type(8)));
typedef float    v8f  __attribute__((ext_vector_type(8)));

// CDNA5 async global->LDS copy support (guarded; falls back to direct loads)
#if defined(__HIP_DEVICE_COMPILE__) && defined(__gfx1250__) && defined(__has_builtin)
#if __has_builtin(__builtin_amdgcn_global_load_async_to_lds_b128) && \
    __has_builtin(__builtin_amdgcn_s_wait_asynccnt)
#define HAVE_ASYNC 1
#endif
#endif
#ifndef HAVE_ASYNC
#define HAVE_ASYNC 0
#endif

#if HAVE_ASYNC
// Exact parameter types per hipcc diagnostic: param0 is
// '__attribute__((__vector_size__(4 * sizeof(int)))) int __device__ *'
// i.e. address_space(1) pointer to a gcc-vector int4; param1 is the LDS
// (address_space(3)) counterpart.
typedef int v4i_t __attribute__((__vector_size__(4 * sizeof(int))));
typedef __attribute__((address_space(1))) v4i_t as1_v4i;
typedef __attribute__((address_space(3))) v4i_t as3_v4i;
#endif

// Load a 16-half A/B fragment for v_wmma_f32_16x16x32_f16.
// rowbase points at 32 contiguous f16 (one row of K, or one B column stored
// K-contiguous). Lanes 0-15 hold K {0..7,16..23}, lanes 16-31 K {8..15,24..31}.
__device__ __forceinline__ v16h load_frag(const _Float16* rowbase, int hs) {
  v8h x0 = *(const v8h*)(rowbase + hs * 8);
  v8h x1 = *(const v8h*)(rowbase + 16 + hs * 8);
  return __builtin_shufflevector(x0, x1, 0,1,2,3,4,5,6,7,8,9,10,11,12,13,14,15);
}

__device__ __forceinline__ v8f wmma_f16(v16h a, v16h b, v8f c) {
  return __builtin_amdgcn_wmma_f32_16x16x32_f16(false, a, false, b, (short)0, c,
                                                false, false);
}

// ---------------- K1: x5p = conv_w @ x5 + conv_b + x5 (per-batch GEMM) ------
__global__ __launch_bounds__(128) void k1_conv_res(
    const float* __restrict__ x5, const float* __restrict__ w,
    const float* __restrict__ bias, float* __restrict__ x5p,
    _Float16* __restrict__ x5ph) {
  __shared__ __align__(16) _Float16 As[64 * 32];  // [m][k]
  __shared__ __align__(16) _Float16 Bs[64 * 32];  // [n][k]
  const int tid = threadIdx.x;
  const int n0 = blockIdx.x * 64, m0 = blockIdx.y * 64, b = blockIdx.z;
  const int wave = tid >> 5, lane = tid & 31, hs = lane >> 4, col = lane & 15;
  v8f acc[4];
  v8f zero = {0.f,0.f,0.f,0.f,0.f,0.f,0.f,0.f};
  for (int t = 0; t < 4; ++t) acc[t] = zero;

  for (int ks = 0; ks < CC / 32; ++ks) {
    const int k0 = ks * 32;
    for (int idx = tid; idx < 2048; idx += 128) {
      int k = idx & 31, m = idx >> 5;
      As[m * 32 + k] = (_Float16)w[(size_t)(m0 + m) * CC + k0 + k];
    }
    for (int idx = tid; idx < 2048; idx += 128) {
      int n = idx & 63, k = idx >> 6;
      Bs[n * 32 + k] = (_Float16)x5[((size_t)(b * CC + k0 + k)) * HWp + n0 + n];
    }
    __syncthreads();
    v16h a = load_frag(&As[(wave * 16 + col) * 32], hs);
    for (int nt = 0; nt < 4; ++nt) {
      v16h bf = load_frag(&Bs[(nt * 16 + col) * 32], hs);
      acc[nt] = wmma_f16(a, bf, acc[nt]);
    }
    __syncthreads();
  }
  for (int nt = 0; nt < 4; ++nt) {
    int p = n0 + nt * 16 + col;
    for (int v = 0; v < 8; ++v) {
      int o = m0 + wave * 16 + v + 8 * hs;
      size_t off = ((size_t)(b * CC + o)) * HWp + p;
      float val = acc[nt][v] + bias[o] + x5[off];
      x5p[off] = val;
      x5ph[off] = (_Float16)val;
    }
  }
}

// ---------------- K2: q/k GEMMs; both stored as [pixel][channel] f16 --------
__global__ __launch_bounds__(128) void k2_qk(
    const _Float16* __restrict__ x5ph, const float* __restrict__ qw,
    const float* __restrict__ qb, const float* __restrict__ kw,
    const float* __restrict__ kbias, _Float16* __restrict__ qh,
    _Float16* __restrict__ khT) {
  __shared__ __align__(16) _Float16 As[64 * 32];
  __shared__ __align__(16) _Float16 Bs[64 * 32];
  const int tid = threadIdx.x;
  const int bz = blockIdx.z, which = bz >> 3, b = bz & 7;
  const float* w = which ? kw : qw;
  const float* bias = which ? kbias : qb;
  _Float16* outT = which ? khT : qh;
  const int n0 = blockIdx.x * 64, m0 = blockIdx.y * 64;
  const int wave = tid >> 5, lane = tid & 31, hs = lane >> 4, col = lane & 15;
  v8f acc[4];
  v8f zero = {0.f,0.f,0.f,0.f,0.f,0.f,0.f,0.f};
  for (int t = 0; t < 4; ++t) acc[t] = zero;

  for (int ks = 0; ks < CC / 32; ++ks) {
    const int k0 = ks * 32;
    for (int idx = tid; idx < 2048; idx += 128) {
      int k = idx & 31, m = idx >> 5;
      As[m * 32 + k] = (_Float16)w[(size_t)(m0 + m) * CC + k0 + k];
    }
    for (int idx = tid; idx < 2048; idx += 128) {
      int n = idx & 63, k = idx >> 6;
      Bs[n * 32 + k] = x5ph[((size_t)(b * CC + k0 + k)) * HWp + n0 + n];
    }
    __syncthreads();
    v16h a = load_frag(&As[(wave * 16 + col) * 32], hs);
    for (int nt = 0; nt < 4; ++nt) {
      v16h bf = load_frag(&Bs[(nt * 16 + col) * 32], hs);
      acc[nt] = wmma_f16(a, bf, acc[nt]);
    }
    __syncthreads();
  }
  for (int nt = 0; nt < 4; ++nt) {
    int p = n0 + nt * 16 + col;
    for (int v = 0; v < 8; ++v) {
      int o = m0 + wave * 16 + v + 8 * hs;
      float val = acc[nt][v] + bias[o];
      outT[((size_t)(b * HWp + p)) * CC + o] = (_Float16)val;
    }
  }
}

// ---------------- K3: fused attention + block-max ---------------------------
// Block = 64 query rows x one key batch (1024 keys), 4 waves.
// Each B fragment is reused across 4 M-tiles (4 WMMAs per B load).
// B tiles are double-buffered into LDS with async global->LDS copies when
// the toolchain exposes them; otherwise loaded directly from global.
#if HAVE_ASYNC
__device__ __forceinline__ void stage_b_async(const _Float16* __restrict__ khT,
                                              _Float16* buf, int jbase,
                                              int lane) {
  // copy 16 rows x 512 halves (16KB): 32 async b128 lane-transfers per instr
  for (int i = 0; i < 32; ++i) {
    int chunk = i * 32 + lane;
    int r = chunk >> 6, c = (chunk & 63) * 8;
    const _Float16* gp = khT + (size_t)(jbase + r) * CC + c;
    _Float16* lp = buf + r * CC + c;
    __builtin_amdgcn_global_load_async_to_lds_b128((as1_v4i*)gp, (as3_v4i*)lp,
                                                   0, 0);
  }
}
#endif

__global__ __launch_bounds__(128) void k3_attmax(
    const _Float16* __restrict__ qh, const _Float16* __restrict__ khT,
    float* __restrict__ maxpart) {
  extern __shared__ __align__(16) char smem[];
  _Float16* Aq = (_Float16*)smem;                       // 64 x 512 f16 = 64 KB
  __shared__ float red[4 * 64];
  const int tid = threadIdx.x;
  const int qt0 = blockIdx.x * 64;  // 128 blocks over BHW
  const int kb = blockIdx.y;        // 0..7
  const int wave = tid >> 5, lane = tid & 31, hs = lane >> 4, col = lane & 15;

  for (int idx = tid; idx < 4096; idx += 128) {
    int c8 = (idx & 63) * 8, r = idx >> 6;
    *(v8h*)&Aq[r * CC + c8] = *(const v8h*)&qh[((size_t)(qt0 + r)) * CC + c8];
  }
  __syncthreads();

  float rmax[4][8];
  for (int mt = 0; mt < 4; ++mt)
    for (int v = 0; v < 8; ++v) rmax[mt][v] = -3.0e38f;

#if HAVE_ASYNC
  _Float16* Bbuf = (_Float16*)(smem + 64 * CC * 2);     // 4 waves x 2 x 16KB
  _Float16* buf0 = Bbuf + (size_t)wave * 2 * 16 * CC;
  _Float16* buf1 = buf0 + 16 * CC;
  stage_b_async(khT, buf0, kb * HWp + wave * 16, lane);
  int cur = 0;
#endif

  for (int nt = wave; nt < 64; nt += 4) {
#if HAVE_ASYNC
    _Float16* bufc = cur ? buf1 : buf0;
    _Float16* bufn = cur ? buf0 : buf1;
    if (nt + 4 < 64) {
      stage_b_async(khT, bufn, kb * HWp + (nt + 4) * 16, lane);
      __builtin_amdgcn_s_wait_asynccnt(32);  // current tile landed in LDS
    } else {
      __builtin_amdgcn_s_wait_asynccnt(0);
    }
    const _Float16* brow = bufc + col * CC;
    cur ^= 1;
#else
    const _Float16* brow = &khT[(size_t)(kb * HWp + nt * 16 + col) * CC];
    if (nt + 4 < 64)
      __builtin_prefetch(&khT[(size_t)(kb * HWp + (nt + 4) * 16 + col) * CC], 0, 0);
#endif
    v8f acc[4];
    v8f zero = {0.f,0.f,0.f,0.f,0.f,0.f,0.f,0.f};
    for (int mt = 0; mt < 4; ++mt) acc[mt] = zero;
    for (int ks = 0; ks < CC / 32; ++ks) {
      v16h bf = load_frag(brow + ks * 32, hs);
      for (int mt = 0; mt < 4; ++mt) {
        v16h a = load_frag(&Aq[(mt * 16 + col) * CC + ks * 32], hs);
        acc[mt] = wmma_f16(a, bf, acc[mt]);
      }
    }
    for (int mt = 0; mt < 4; ++mt)
      for (int v = 0; v < 8; ++v) {
        float t = acc[mt][v];
        for (int m1 = 1; m1 < 16; m1 <<= 1) t = fmaxf(t, __shfl_xor(t, m1, 16));
        rmax[mt][v] = fmaxf(rmax[mt][v], t);
      }
  }
  if (col == 0)
    for (int mt = 0; mt < 4; ++mt)
      for (int v = 0; v < 8; ++v)
        red[wave * 64 + mt * 16 + v + 8 * hs] = rmax[mt][v];
  __syncthreads();
  if (tid < 64) {
    float mx = fmaxf(fmaxf(red[tid], red[64 + tid]),
                     fmaxf(red[128 + tid], red[192 + tid]));
    maxpart[(size_t)(qt0 + tid) * 8 + kb] = mx;
  }
}

// ---------------- K4: mean over key batches + softmax per batch -------------
__global__ __launch_bounds__(256) void k4_softmax(
    const float* __restrict__ maxpart, float* __restrict__ xw,
    float* __restrict__ xwmax) {
  __shared__ float sv[HWp];
  __shared__ float red[256];
  const int b = blockIdx.x, tid = threadIdx.x;
  const float sc = (1.0f / sqrtf(512.0f)) / 8.0f;
  for (int p = tid; p < HWp; p += 256) {
    const float* mp = &maxpart[(size_t)(b * HWp + p) * 8];
    float s = 0.f;
    for (int k = 0; k < 8; ++k) s += mp[k];
    sv[p] = s * sc;
  }
  __syncthreads();
  float m = -3e38f;
  for (int p = tid; p < HWp; p += 256) m = fmaxf(m, sv[p]);
  red[tid] = m; __syncthreads();
  for (int s = 128; s > 0; s >>= 1) {
    if (tid < s) red[tid] = fmaxf(red[tid], red[tid + s]);
    __syncthreads();
  }
  float vmax = red[0];
  __syncthreads();
  float ssum = 0.f;
  for (int p = tid; p < HWp; p += 256) {
    float e = __expf(sv[p] - vmax);
    sv[p] = e; ssum += e;
  }
  red[tid] = ssum; __syncthreads();
  for (int s = 128; s > 0; s >>= 1) {
    if (tid < s) red[tid] += red[tid + s];
    __syncthreads();
  }
  float inv = 1.0f / red[0];
  for (int p = tid; p < HWp; p += 256) xw[b * HWp + p] = sv[p] * inv;
  if (tid == 0) xwmax[b] = inv;  // softmax value at argmax = exp(0)/sum
}

// ---------------- K5: L2-norm inverses + masked seed vectors ----------------
__global__ __launch_bounds__(256) void k5_seeds(
    const float* __restrict__ x5p, const float* __restrict__ xw,
    const float* __restrict__ xwmax, float* __restrict__ invn,
    float* __restrict__ seeds) {
  __shared__ float sseed[CC];
  const int b = blockIdx.x, tid = threadIdx.x;
  for (int c = tid; c < CC; c += 256) sseed[c] = 0.0f;
  __syncthreads();
  float mx = xwmax[b];
  for (int p = tid; p < HWp; p += 256) {
    const float* xp = &x5p[(size_t)b * CC * HWp + p];
    float ss = 0.f;
    for (int c = 0; c < CC; ++c) { float v = xp[(size_t)c * HWp]; ss += v * v; }
    float inv = 1.0f / fmaxf(sqrtf(ss), 1e-12f);
    invn[b * HWp + p] = inv;
    if (xw[b * HWp + p] == mx)
      for (int c = 0; c < CC; ++c)
        atomicAdd(&sseed[c], xp[(size_t)c * HWp] * inv);
  }
  __syncthreads();
  for (int c = tid; c < CC; c += 256) seeds[b * CC + c] = sseed[c];
}

// ---------------- K6: correlation maps, relu, mean over 8 seeds -------------
__global__ __launch_bounds__(256) void k6_cm(
    const float* __restrict__ x5p, const float* __restrict__ invn,
    const float* __restrict__ seeds, float* __restrict__ cm) {
  const int i = blockIdx.x * 256 + threadIdx.x;  // 0..8191
  const int b = i >> 10, p = i & 1023;
  float inv = invn[i];
  float acc[8];
  for (int k = 0; k < 8; ++k) acc[k] = 0.f;
  const float* xp = &x5p[(size_t)b * CC * HWp + p];
  for (int c = 0; c < CC; ++c) {
    float v = xp[(size_t)c * HWp] * inv;
    for (int k = 0; k < 8; ++k) acc[k] = fmaf(v, seeds[k * CC + c], acc[k]);
  }
  float s = 0.f;
  for (int k = 0; k < 8; ++k) s += fmaxf(acc[k], 0.f);
  cm[i] = s * 0.125f;
}

// ---------------- K7: per-batch min/max normalize → cormap ------------------
__global__ __launch_bounds__(256) void k7_cormap(
    const float* __restrict__ cm, float* __restrict__ cormap) {
  __shared__ float rmn[256], rmx[256];
  const int b = blockIdx.x, tid = threadIdx.x;
  float mn = 3e38f, mx = -3e38f;
  for (int p = tid; p < HWp; p += 256) {
    float v = cm[b * HWp + p];
    mn = fminf(mn, v); mx = fmaxf(mx, v);
  }
  rmn[tid] = mn; rmx[tid] = mx; __syncthreads();
  for (int s = 128; s > 0; s >>= 1) {
    if (tid < s) {
      rmn[tid] = fminf(rmn[tid], rmn[tid + s]);
      rmx[tid] = fmaxf(rmx[tid], rmx[tid + s]);
    }
    __syncthreads();
  }
  mn = rmn[0]; mx = rmx[0];
  float inv = 1.0f / (mx - mn + 1e-12f);
  for (int p = tid; p < HWp; p += 256)
    cormap[b * HWp + p] = (cm[b * HWp + p] - mn) * inv;
}

// ---------------- K8a: proto1[c] = mean over (b,hw) of x5p*cormap -----------
__global__ __launch_bounds__(256) void k8_proto(
    const float* __restrict__ x5p, const float* __restrict__ cormap,
    float* __restrict__ out_proto) {
  __shared__ float red[256];
  const int c = blockIdx.x, tid = threadIdx.x;
  float s = 0.f;
  for (int j = tid; j < BHW; j += 256) {
    int b = j >> 10, p = j & 1023;
    s += x5p[((size_t)(b * CC + c)) * HWp + p] * cormap[j];
  }
  red[tid] = s; __syncthreads();
  for (int st = 128; st > 0; st >>= 1) {
    if (tid < st) red[tid] += red[tid + st];
    __syncthreads();
  }
  if (tid == 0) out_proto[c] = red[0] * (1.0f / (float)BHW);
}

// ---------------- K8b: out = x5p*proto + x5p*cormap -------------------------
__global__ __launch_bounds__(256) void k8_final(
    const float* __restrict__ x5p, const float* __restrict__ cormap,
    const float* __restrict__ proto, float* __restrict__ out) {
  size_t idx = (size_t)blockIdx.x * 256 + threadIdx.x;
  int p = (int)(idx & 1023);
  int c = (int)((idx >> 10) & 511);
  int b = (int)(idx >> 19);
  float v = x5p[idx];
  out[idx] = v * proto[c] + v * cormap[b * HWp + p];
}

extern "C" void kernel_launch(void* const* d_in, const int* in_sizes, int n_in,
                              void* d_out, int out_size, void* d_ws,
                              size_t ws_size, hipStream_t stream) {
  const float* x5     = (const float*)d_in[0];
  const float* conv_w = (const float*)d_in[1];
  const float* conv_b = (const float*)d_in[2];
  const float* q_w    = (const float*)d_in[3];
  const float* q_b    = (const float*)d_in[4];
  const float* k_w    = (const float*)d_in[5];
  const float* k_b    = (const float*)d_in[6];

  char* ws = (char*)d_ws;
  size_t off = 0;
  auto carve = [&](size_t bytes) -> void* {
    void* p = ws + off;
    off += (bytes + 255) & ~(size_t)255;
    return p;
  };
  float*    x5p     = (float*)carve(sizeof(float) * BB * CC * HWp);
  _Float16* x5ph    = (_Float16*)carve(sizeof(_Float16) * BB * CC * HWp);
  _Float16* qh      = (_Float16*)carve(sizeof(_Float16) * BHW * CC);
  _Float16* khT     = (_Float16*)carve(sizeof(_Float16) * BHW * CC);
  float*    maxpart = (float*)carve(sizeof(float) * BHW * 8);
  float*    xw      = (float*)carve(sizeof(float) * BB * HWp);
  float*    xwmax   = (float*)carve(sizeof(float) * BB);
  float*    invn    = (float*)carve(sizeof(float) * BB * HWp);
  float*    seeds   = (float*)carve(sizeof(float) * BB * CC);
  float*    cm      = (float*)carve(sizeof(float) * BB * HWp);
  float*    cormap  = (float*)carve(sizeof(float) * BB * HWp);

  float* out_proto = (float*)d_out;        // [1,C,1,1] = 512 floats
  float* out_main  = (float*)d_out + CC;   // [B,C,H,W]

  // K3 dynamic LDS: 64x512 f16 A-strip + 4 waves x 2 x (16x512 f16) B buffers
  const size_t k3_smem = (size_t)64 * CC * 2 + (size_t)4 * 2 * 16 * CC * 2;

  k1_conv_res<<<dim3(HWp / 64, CC / 64, BB), 128, 0, stream>>>(
      x5, conv_w, conv_b, x5p, x5ph);
  k2_qk<<<dim3(HWp / 64, CC / 64, 2 * BB), 128, 0, stream>>>(
      x5ph, q_w, q_b, k_w, k_b, qh, khT);
  k3_attmax<<<dim3(BHW / 64, BB), 128, k3_smem, stream>>>(qh, khT, maxpart);
  k4_softmax<<<dim3(BB), 256, 0, stream>>>(maxpart, xw, xwmax);
  k5_seeds<<<dim3(BB), 256, 0, stream>>>(x5p, xw, xwmax, invn, seeds);
  k6_cm<<<dim3(BHW / 256), 256, 0, stream>>>(x5p, invn, seeds, cm);
  k7_cormap<<<dim3(BB), 256, 0, stream>>>(cm, cormap);
  k8_proto<<<dim3(CC), 256, 0, stream>>>(x5p, cormap, out_proto);
  k8_final<<<dim3((BB * CC * HWp) / 256), 256, 0, stream>>>(
      x5p, cormap, out_proto, out_main);
}
